// SDNEncoder_40149354283110
// MI455X (gfx1250) — matrix operand order repaired and checked by reading, not compiled, not run on hardware
//
#include <hip/hip_runtime.h>

typedef __bf16 bf16;
typedef __attribute__((ext_vector_type(16))) __bf16 v16bf;
typedef __attribute__((ext_vector_type(8)))  __bf16 v8bf;
typedef __attribute__((ext_vector_type(8)))  float   v8f;

#define NNODE  100
#define WIN    128
#define GHC    256
#define DHC    64
#define OUTC   128
#define BATCH  256
#define NTOT   (BATCH * NNODE)     // 25600
#define NEDGE  409600
#define FEAT   768                 // 6 * WIN

// ---------------- f32 -> bf16 convert ----------------
__global__ void k_cvt_bf16(const float* __restrict__ src, bf16* __restrict__ dst, int count) {
    int t = blockIdx.x * blockDim.x + threadIdx.x;
    if (t < count) dst[t] = (bf16)src[t];
}

// ---------------- multiscale conv + relu -> bf16 [B, n, 768] ----------------
__global__ void k_conv(const float* __restrict__ x,
                       const float* __restrict__ w3, const float* __restrict__ b3,
                       const float* __restrict__ w5, const float* __restrict__ b5,
                       const float* __restrict__ w7, const float* __restrict__ b7,
                       bf16* __restrict__ cbf) {
    int t = blockIdx.x * blockDim.x + threadIdx.x;   // NTOT * WIN threads
    if (t >= NTOT * WIN) return;
    int w = t & (WIN - 1);
    int r = t >> 7;                 // flat node row 0..25599 ( = b*100 + i )
    int i = r % NNODE;
    float xv[7];
#pragma unroll
    for (int d = 0; d < 7; ++d) {
        int p = w + d - 3;
        xv[d] = (p >= 0 && p < WIN) ? x[(size_t)r * WIN + p] : 0.f;
    }
    bf16* out = cbf + (size_t)r * FEAT + w;
#pragma unroll
    for (int c = 0; c < 2; ++c) {
        float a = b3[i * 2 + c];
#pragma unroll
        for (int j = 0; j < 3; ++j) a += xv[2 + j] * w3[(i * 2 + c) * 3 + j];
        out[(0 + c) * WIN] = (bf16)fmaxf(a, 0.f);

        float b = b5[i * 2 + c];
#pragma unroll
        for (int j = 0; j < 5; ++j) b += xv[1 + j] * w5[(i * 2 + c) * 5 + j];
        out[(2 + c) * WIN] = (bf16)fmaxf(b, 0.f);

        float d7 = b7[i * 2 + c];
#pragma unroll
        for (int j = 0; j < 7; ++j) d7 += xv[j] * w7[(i * 2 + c) * 7 + j];
        out[(4 + c) * WIN] = (bf16)fmaxf(d7, 0.f);
    }
}

// ---------------- generic bf16 WMMA GEMM: D = relu?(A * Bt^T + bias) ----------------
// A: [M x K] row-major (row stride lda), per-node base offset aNode.
// Bt: [N x K] row-major (= column-major B), row stride ldb, per-node base bNode.
// One wave computes one 16x16 tile; tilesPerNode is a multiple of 8 (waves/block).
__global__ void k_gemm_wmma(const bf16* __restrict__ A, long long aNode, int lda,
                            const bf16* __restrict__ Bt, long long bNode, int ldb,
                            const float* __restrict__ bias, int biasNode,
                            float* __restrict__ Df, bf16* __restrict__ Dbf,
                            long long dNode, int ldd,
                            int K, int ntiles, int tilesPerNode, int relu) {
    int t = blockIdx.x * 8 + (threadIdx.x >> 5);   // wave index within node
    if (t >= tilesPerNode) return;                 // wave-uniform
    int node = blockIdx.z;
    int mt = t / ntiles, nt = t % ntiles;
    int lane = threadIdx.x & 31;
    int half = lane >> 4;          // 0: K-lo lanes, 1: K-hi lanes
    int l15  = lane & 15;

    const bf16* Arow = A + (long long)node * aNode + (long long)(mt * 16 + l15) * lda + half * 16;
    const bf16* Bcol = Bt + (long long)node * bNode + (long long)(nt * 16 + l15) * ldb + half * 16;

    v8f acc = {0.f, 0.f, 0.f, 0.f, 0.f, 0.f, 0.f, 0.f};
    for (int k = 0; k < K; k += 32) {
        v8bf alo = *(const v8bf*)(Arow + k);        // K = k + half*16 + [0..7]
        v8bf ahi = *(const v8bf*)(Arow + k + 8);    // K = k + half*16 + 8 + [0..7]
        v16bf bfrag = *(const v16bf*)(Bcol + k);    // K = k + half*16 + [0..15]
        v16bf afrag;
#pragma unroll
        for (int j = 0; j < 8; ++j) { afrag[j] = alo[j]; afrag[j + 8] = ahi[j]; }
        acc = __builtin_amdgcn_wmma_f32_16x16x32_bf16(false, afrag, false, bfrag,
                                                      (short)0, acc, false, false);
    }

    int ncol = nt * 16 + l15;
    float bv = bias ? bias[node * biasNode + ncol] : 0.f;
    int mbase = mt * 16 + half * 8;                // D: VGPR r -> M = r + 8*(lane/16)
#pragma unroll
    for (int r = 0; r < 8; ++r) {
        float v = acc[r] + bv;
        if (relu) v = fmaxf(v, 0.f);
        long long idx = (long long)node * dNode + (long long)(mbase + r) * ldd + ncol;
        if (Df) Df[idx] = v;
        else    Dbf[idx] = (bf16)v;
    }
}

// ---------------- GCN helpers ----------------
__global__ void k_deg_init(float* d1, float* d2) {
    int t = blockIdx.x * blockDim.x + threadIdx.x;
    if (t < NTOT) { d1[t] = 1.f; d2[t] = 1.f; }    // self-loop contributes 1
}
__global__ void k_edge_deg(const int* __restrict__ ei, float* __restrict__ deg) {
    int e = blockIdx.x * blockDim.x + threadIdx.x;
    if (e < NEDGE) {
        int d = ei[NEDGE + e];
        atomicAdd(&deg[d], 1.f);
    }
}
__global__ void k_norm(float* d1, float* d2) {
    int t = blockIdx.x * blockDim.x + threadIdx.x;
    if (t < NTOT) {
        d1[t] = rsqrtf(fmaxf(d1[t], 1e-12f));
        d2[t] = rsqrtf(fmaxf(d2[t], 1e-12f));
    }
}
// agg init folds the self-loop: agg[r] = norm[r]^2 * xw[r]
__global__ void k_agg_init(const float* __restrict__ xw, const float* __restrict__ nrm,
                           float* __restrict__ agg) {
    size_t t = (size_t)blockIdx.x * blockDim.x + threadIdx.x;
    if (t >= (size_t)NTOT * GHC) return;
    int r = (int)(t >> 8);
    float nn = nrm[r];
    agg[t] = nn * nn * xw[t];
}
__global__ void k_scatter(const int* __restrict__ ei, const float* __restrict__ nrm,
                          const float* __restrict__ xw, float* __restrict__ agg) {
    int e = blockIdx.x;            // one block per edge, 256 threads = 256 features
    int f = threadIdx.x;
    int s = ei[e];
    int d = ei[NEDGE + e];
    float coef = nrm[s] * nrm[d];
    atomicAdd(&agg[(size_t)d * GHC + f], coef * xw[(size_t)s * GHC + f]);
}
// cat = [relu(agg1+bg) || relu(agg2+bg)] as bf16, [NTOT, 512]
__global__ void k_cat(const float* __restrict__ agg1, const float* __restrict__ agg2,
                      const float* __restrict__ bg, bf16* __restrict__ cat) {
    size_t t = (size_t)blockIdx.x * blockDim.x + threadIdx.x;
    if (t >= (size_t)NTOT * GHC) return;
    size_t r = t >> 8;
    int f = (int)(t & 255);
    float bv = bg[f];
    cat[r * 512 + f]       = (bf16)fmaxf(agg1[t] + bv, 0.f);
    cat[r * 512 + 256 + f] = (bf16)fmaxf(agg2[t] + bv, 0.f);
}

extern "C" void kernel_launch(void* const* d_in, const int* in_sizes, int n_in,
                              void* d_out, int out_size, void* d_ws, size_t ws_size,
                              hipStream_t stream) {
    (void)in_sizes; (void)n_in; (void)out_size; (void)ws_size;
    const float* x1  = (const float*)d_in[0];
    const float* x2  = (const float*)d_in[1];
    const int*   ei1 = (const int*)d_in[2];
    const int*   ei2 = (const int*)d_in[3];
    const float* w3  = (const float*)d_in[4];
    const float* b3  = (const float*)d_in[5];
    const float* w5  = (const float*)d_in[6];
    const float* b5  = (const float*)d_in[7];
    const float* w7  = (const float*)d_in[8];
    const float* b7  = (const float*)d_in[9];
    const float* Wff = (const float*)d_in[10];
    const float* bff = (const float*)d_in[11];
    const float* Wg  = (const float*)d_in[12];
    const float* bg  = (const float*)d_in[13];
    const float* Wd  = (const float*)d_in[14];
    const float* bd  = (const float*)d_in[15];
    const float* Wfc = (const float*)d_in[16];
    const float* bfc = (const float*)d_in[17];
    float* out = (float*)d_out;

    // ---- workspace layout (256B aligned) ----
    char* base = (char*)d_ws;
    size_t o = 0;
    auto alloc = [&](size_t bytes) { size_t r = o; o += (bytes + 255) & ~(size_t)255; return r; };
    size_t oWffb = alloc((size_t)NNODE * GHC * FEAT * 2);   // 39.3 MB
    size_t oWgb  = alloc((size_t)GHC * GHC * 2);
    size_t oWdb  = alloc((size_t)NNODE * DHC * 2 * GHC * 2);
    size_t oWfcb = alloc((size_t)OUTC * NNODE * DHC * 2);
    size_t oC1   = alloc((size_t)NTOT * FEAT * 2);          // cbf1, later xw1 (f32), later cat
    size_t oC2   = alloc((size_t)NTOT * FEAT * 2);          // cbf2, later xw2 (f32)
    size_t oH1   = alloc((size_t)NTOT * GHC * 2);
    size_t oH2   = alloc((size_t)NTOT * GHC * 2);
    size_t oAg1  = alloc((size_t)NTOT * GHC * 4);
    size_t oAg2  = alloc((size_t)NTOT * GHC * 4);
    size_t oDg1  = alloc((size_t)NTOT * 4);
    size_t oDg2  = alloc((size_t)NTOT * 4);
    size_t oDu   = alloc((size_t)BATCH * NNODE * DHC * 2);

    bf16*  Wffb = (bf16*)(base + oWffb);
    bf16*  Wgb  = (bf16*)(base + oWgb);
    bf16*  Wdb  = (bf16*)(base + oWdb);
    bf16*  Wfcb = (bf16*)(base + oWfcb);
    bf16*  cbf1 = (bf16*)(base + oC1);
    bf16*  cbf2 = (bf16*)(base + oC2);
    float* xw1  = (float*)(base + oC1);    // overlays cbf1 (free after FF1)
    float* xw2  = (float*)(base + oC2);    // overlays cbf2 (free after FF2)
    bf16*  cat  = (bf16*)(base + oC1);     // overlays xw1 (free after both scatters)
    bf16*  h1   = (bf16*)(base + oH1);
    bf16*  h2   = (bf16*)(base + oH2);
    float* agg1 = (float*)(base + oAg1);
    float* agg2 = (float*)(base + oAg2);
    float* deg1 = (float*)(base + oDg1);
    float* deg2 = (float*)(base + oDg2);
    bf16*  dub  = (bf16*)(base + oDu);

    const int TB = 256;
    auto G = [](long long n, int b) { return (unsigned)((n + b - 1) / b); };

    // 1) weight conversions to bf16
    k_cvt_bf16<<<G((long long)NNODE * GHC * FEAT, TB), TB, 0, stream>>>(Wff, Wffb, NNODE * GHC * FEAT);
    k_cvt_bf16<<<G(GHC * GHC, TB), TB, 0, stream>>>(Wg, Wgb, GHC * GHC);
    k_cvt_bf16<<<G((long long)NNODE * DHC * 2 * GHC, TB), TB, 0, stream>>>(Wd, Wdb, NNODE * DHC * 2 * GHC);
    k_cvt_bf16<<<G((long long)OUTC * NNODE * DHC, TB), TB, 0, stream>>>(Wfc, Wfcb, OUTC * NNODE * DHC);

    // 2) multiscale conv -> bf16 activations
    k_conv<<<G((long long)NTOT * WIN, TB), TB, 0, stream>>>(x1, w3, b3, w5, b5, w7, b7, cbf1);
    k_conv<<<G((long long)NTOT * WIN, TB), TB, 0, stream>>>(x2, w3, b3, w5, b5, w7, b7, cbf2);

    // 3) per-node FF GEMMs: h = relu(c @ Wff^T + bff)   [256x768]x[768x256] per node
    {
        dim3 grid(32, 1, NNODE);   // 256 tiles/node, 8 waves/block
        k_gemm_wmma<<<grid, TB, 0, stream>>>(cbf1, 768, NNODE * FEAT,
                                             Wffb, (long long)GHC * FEAT, FEAT,
                                             bff, GHC,
                                             nullptr, h1, GHC, NNODE * GHC,
                                             FEAT, 16, 256, 1);
        k_gemm_wmma<<<grid, TB, 0, stream>>>(cbf2, 768, NNODE * FEAT,
                                             Wffb, (long long)GHC * FEAT, FEAT,
                                             bff, GHC,
                                             nullptr, h2, GHC, NNODE * GHC,
                                             FEAT, 16, 256, 1);
    }

    // 4) GCN linear: xw = h @ Wg^T   [25600x256]x[256x256]
    {
        dim3 grid(3200, 1, 1);     // 25600 tiles
        k_gemm_wmma<<<grid, TB, 0, stream>>>(h1, 0, GHC, Wgb, 0, GHC,
                                             nullptr, 0, xw1, nullptr, 0, GHC,
                                             GHC, 16, 25600, 0);
        k_gemm_wmma<<<grid, TB, 0, stream>>>(h2, 0, GHC, Wgb, 0, GHC,
                                             nullptr, 0, xw2, nullptr, 0, GHC,
                                             GHC, 16, 25600, 0);
    }

    // 5) degrees (self-loop + incoming edges) -> symmetric norm
    k_deg_init<<<G(NTOT, TB), TB, 0, stream>>>(deg1, deg2);
    k_edge_deg<<<G(NEDGE, TB), TB, 0, stream>>>(ei1, deg1);
    k_edge_deg<<<G(NEDGE, TB), TB, 0, stream>>>(ei2, deg2);
    k_norm<<<G(NTOT, TB), TB, 0, stream>>>(deg1, deg2);

    // 6) aggregation: init with self-loop term, then edge scatter (f32 atomics; xw is L2-resident)
    k_agg_init<<<G((long long)NTOT * GHC, TB), TB, 0, stream>>>(xw1, deg1, agg1);
    k_agg_init<<<G((long long)NTOT * GHC, TB), TB, 0, stream>>>(xw2, deg2, agg2);
    k_scatter<<<NEDGE, GHC, 0, stream>>>(ei1, deg1, xw1, agg1);
    k_scatter<<<NEDGE, GHC, 0, stream>>>(ei2, deg2, xw2, agg2);

    // 7) bias + relu + concat -> bf16 [NTOT, 512]  (overlays xw1 region; safe, stream-ordered)
    k_cat<<<G((long long)NTOT * GHC, TB), TB, 0, stream>>>(agg1, agg2, bg, cat);

    // 8) per-node discrepancy FC: du = relu(cat @ Wd^T + bd) -> bf16 [256, 6400] node-major
    {
        dim3 grid(8, 1, NNODE);    // 64 tiles/node
        k_gemm_wmma<<<grid, TB, 0, stream>>>(cat, 512, NNODE * 512,
                                             Wdb, (long long)DHC * 2 * GHC, 2 * GHC,
                                             bd, DHC,
                                             nullptr, dub, DHC, NNODE * DHC,
                                             2 * GHC, 4, 64, 1);
    }

    // 9) final FC: out = du @ Wfc^T + bfc   [256x6400]x[6400x128]
    {
        dim3 grid(16, 1, 1);       // 128 tiles
        k_gemm_wmma<<<grid, TB, 0, stream>>>(dub, 0, NNODE * DHC,
                                             Wfcb, 0, NNODE * DHC,
                                             bfc, 0,
                                             out, nullptr, 0, OUTC,
                                             NNODE * DHC, 8, 128, 0);
    }
}
